// GCN_77111842832773
// MI455X (gfx1250) — matrix operand order repaired and checked by reading, not compiled, not run on hardware
//
#include <hip/hip_runtime.h>
#include <hip/hip_bf16.h>

// ---------------------------------------------------------------------------
// GraphSAGE (3x SAGEConv + BN + ReLU, max-pool, linear) for MI455X (gfx1250).
// Project-then-aggregate, bf16 WMMA GEMMs with fp32 accumulation. Weights and
// inter-layer activations pre-packed bf16 so GEMM staging is copy-only, done
// with gfx1250 async global->LDS DMA (ASYNCcnt) when available. Double-
// buffered LDS pipeline; atomics for segment ops. Memory-bound overall.
// ---------------------------------------------------------------------------

#define N_NODES  50000
#define N_EDGES  400000
#define N_GRAPHS 128
#define IN_DIM   768
#define HID      128
#define N_CLS    11
#define BN_EPS   1e-5f

#if __has_builtin(__builtin_amdgcn_global_load_async_to_lds_b128) && \
    __has_builtin(__builtin_amdgcn_s_wait_asynccnt)
#define HAVE_ASYNC_LDS 1
#else
#define HAVE_ASYNC_LDS 0
#endif

typedef __attribute__((ext_vector_type(16))) __bf16 v16bf;
typedef __attribute__((ext_vector_type(8)))  float  v8f;
typedef __attribute__((ext_vector_type(4)))  int    v4i;

typedef __attribute__((address_space(1))) v4i* gv4i_ptr;
typedef __attribute__((address_space(3))) v4i* lv4i_ptr;

union Frag {
    v16bf v;
    uint4 q[2];
};

// fp32 x2 -> packed bf16 dword (round-half-up; 2 adds + 1 perm).
__device__ __forceinline__ unsigned int pk2(float a, float b) {
    unsigned int ua = __float_as_uint(a) + 0x8000u;
    unsigned int ub = __float_as_uint(b) + 0x8000u;
#if __has_builtin(__builtin_amdgcn_perm)
    return __builtin_amdgcn_perm(ub, ua, 0x07060302u);
#else
    return (ua >> 16) | (ub & 0xFFFF0000u);
#endif
}

// async global -> LDS copy of 16 bytes (no VGPR round-trip), ASYNCcnt-tracked.
__device__ __forceinline__ void async_copy_b128(const unsigned int* g, unsigned int* l) {
#if HAVE_ASYNC_LDS
    __builtin_amdgcn_global_load_async_to_lds_b128(
        (gv4i_ptr)(g), (lv4i_ptr)(l), 0, 0);
#endif
}

// ---------------------------------------------------------------------------
// Dual-weight GEMM: C[N x 256] = A[N x K] @ Wcat^T, Wcat = bf16 [256 x K].
// Workgroup: 256 threads = 8 waves; tile = 64 rows x 256 cols.
// Wave (w): rows [(w&3)*16, +16), cols [(w>>2)*128, +128) -> 8 16x16 accums.
// ABF16: A is packed bf16 [N x K/2 dwords]; else fp32 [N x K].
// ---------------------------------------------------------------------------
template <bool ABF16>
__global__ __launch_bounds__(256, 1)
void gemm_dual(const void* __restrict__ Av,
               const unsigned int* __restrict__ Wcat,   // [256][K/2] bf16 dwords
               float* __restrict__ C,
               int N, int K)
{
    // padded LDS (stride 20 dwords = 80B); rows stay 16B-aligned.
    __shared__ unsigned int As[2][64][20];    // 64 rows x 32 bf16 each buf
    __shared__ unsigned int Bs[2][256][20];   // 256 out-dims x 32 bf16 each buf

    const int tid  = threadIdx.x;
    const int wave = tid >> 5;
    const int lane = tid & 31;
    const int rowbase = blockIdx.x * 64;

    const int wm = (wave & 3) * 16;      // wave row offset in tile
    const int wn = (wave >> 2) * 128;    // wave col offset (0 or 128)

    v8f acc[8] = {};

    const int Kd = K >> 1;               // dwords per packed bf16 row
    const int ar = tid >> 2;             // staging row 0..63
    const int aq = tid & 3;              // quarter of 32-wide K slab
    int agrow = rowbase + ar;
    if (agrow >= N) agrow = N - 1;       // clamp; garbage rows masked at store

    const float*        arow_f = (const float*)Av + (size_t)agrow * K + aq * 8;
    const unsigned int* arow_u = (const unsigned int*)Av + (size_t)agrow * Kd + aq * 4;
    const unsigned int* wrow   = Wcat + (size_t)tid * Kd;   // out-dim = tid

    const int nk = K >> 5;
    float4 areg[2];

#if HAVE_ASYNC_LDS
    constexpr short NIN = ABF16 ? 5 : 4;  // async instrs per wave per slab
    // ---- prologue: async-issue slab 0 ----
#pragma unroll
    for (int j = 0; j < 4; ++j)
        async_copy_b128(wrow + j * 4, &Bs[0][tid][j * 4]);
    if (ABF16) {
        async_copy_b128(arow_u, &As[0][ar][aq * 4]);
    } else {
        areg[0] = *(const float4*)(arow_f);
        areg[1] = *(const float4*)(arow_f + 4);
        unsigned int* d = &As[0][ar][aq * 4];
        d[0] = pk2(areg[0].x, areg[0].y);
        d[1] = pk2(areg[0].z, areg[0].w);
        d[2] = pk2(areg[1].x, areg[1].y);
        d[3] = pk2(areg[1].z, areg[1].w);
    }
#else
    uint4 a16;
    uint4 breg[4];
    if (ABF16) {
        a16 = *(const uint4*)(arow_u);
    } else {
        areg[0] = *(const float4*)(arow_f);
        areg[1] = *(const float4*)(arow_f + 4);
    }
#pragma unroll
    for (int j = 0; j < 4; ++j)
        breg[j] = *(const uint4*)(wrow + j * 4);
#endif

    for (int kt = 0; kt < nk; ++kt) {
        const int buf  = kt & 1;
        const int nbuf = buf ^ 1;
        const bool more = (kt + 1 < nk);
        const int kd = (kt + 1) << 4;              // next slab, dword offset

#if HAVE_ASYNC_LDS
        // ---- issue async staging of slab kt+1 into the other buffer ----
        if (more) {
#pragma unroll
            for (int j = 0; j < 4; ++j)
                async_copy_b128(wrow + kd + j * 4, &Bs[nbuf][tid][j * 4]);
            if (ABF16) {
                async_copy_b128(arow_u + kd, &As[nbuf][ar][aq * 4]);
            } else {
                const int kf = (kt + 1) << 5;
                areg[0] = *(const float4*)(arow_f + kf);
                areg[1] = *(const float4*)(arow_f + kf + 4);
            }
        }
        // wait for slab kt only (slab kt+1 stays in flight behind the WMMAs)
        if (more) __builtin_amdgcn_s_wait_asynccnt(NIN);
        else      __builtin_amdgcn_s_wait_asynccnt(0);
        __syncthreads();
#else
        // ---- store previously prefetched slab kt to LDS ----
        {
            unsigned int* d = &As[buf][ar][aq * 4];
            if (ABF16) {
                *(uint4*)d = a16;
            } else {
                d[0] = pk2(areg[0].x, areg[0].y);
                d[1] = pk2(areg[0].z, areg[0].w);
                d[2] = pk2(areg[1].x, areg[1].y);
                d[3] = pk2(areg[1].z, areg[1].w);
            }
            unsigned int* e = &Bs[buf][tid][0];
#pragma unroll
            for (int j = 0; j < 4; ++j)
                *(uint4*)(e + j * 4) = breg[j];
        }
        __syncthreads();
#endif

        // ---- A fragment (16x32 bf16, ISA 7.12.2 layout) ----
        const int arw  = wm + (lane & 15);
        const int koff = (lane >> 4) * 4;          // dword offset (8 bf16)
        Frag af;
        af.q[0] = *(const uint4*)&As[buf][arw][koff];
        af.q[1] = *(const uint4*)&As[buf][arw][8 + koff];

        // ---- 8 column tiles of 16 ----
#pragma unroll
        for (int t = 0; t < 8; ++t) {
            const int nn  = wn + t * 16 + (lane & 15);
            const int bko = (lane >> 4) * 4;       // K 8h..8h+7 and 16+8h..16+8h+7
            Frag bfg;
            bfg.q[0] = *(const uint4*)&Bs[buf][nn][bko];
            bfg.q[1] = *(const uint4*)&Bs[buf][nn][bko + 8];
            acc[t] = __builtin_amdgcn_wmma_f32_16x16x32_bf16(
                false, af.v, false, bfg.v, (short)0, acc[t], false, false);
        }

#if HAVE_ASYNC_LDS
        // fp32-A: convert + store next A slab after compute (next buffer)
        if (!ABF16 && more) {
            unsigned int* d = &As[nbuf][ar][aq * 4];
            d[0] = pk2(areg[0].x, areg[0].y);
            d[1] = pk2(areg[0].z, areg[0].w);
            d[2] = pk2(areg[1].x, areg[1].y);
            d[3] = pk2(areg[1].z, areg[1].w);
        }
        __syncthreads();   // WAR guard: nbuf safe to overwrite next iteration
#else
        // ---- prefetch slab kt+1 AFTER compute: live range spans only the
        // back-edge, not the WMMA pressure peak (avoids scratch spills) ----
        if (more) {
            if (ABF16) {
                a16 = *(const uint4*)(arow_u + kd);
            } else {
                const int kf = (kt + 1) << 5;
                areg[0] = *(const float4*)(arow_f + kf);
                areg[1] = *(const float4*)(arow_f + kf + 4);
            }
#pragma unroll
            for (int j = 0; j < 4; ++j)
                breg[j] = *(const uint4*)(wrow + kd + j * 4);
        }
#endif
    }

    // ---- epilogue: C/D layout: lane&15 = col, lanes>=16 hold M+8, vgpr r = M ----
    const int colb = wn + (lane & 15);
    const int rowb = rowbase + wm + ((lane >> 4) ? 8 : 0);
#pragma unroll
    for (int t = 0; t < 8; ++t) {
        const int col = colb + t * 16;
#pragma unroll
        for (int r = 0; r < 8; ++r) {
            const int row = rowb + r;
            if (row < N) C[(size_t)row * 256 + col] = acc[t][r];
        }
    }
}

// Pack [Wl|Wr] (each 128 x K fp32 row-major) into Wcat bf16 [256][K/2 dwords].
__global__ void convert_w(const float* __restrict__ Wl,
                          const float* __restrict__ Wr,
                          unsigned int* __restrict__ out, int K)
{
    int t = blockIdx.x * blockDim.x + threadIdx.x;
    const int Kd = K >> 1;
    if (t >= 256 * Kd) return;
    int row = t / Kd, cd = t - row * Kd;
    const float* w = ((row < 128) ? (Wl + (size_t)row * K)
                                  : (Wr + (size_t)(row - 128) * K)) + cd * 2;
    out[t] = pk2(w[0], w[1]);
}

// ---------------------------------------------------------------------------
__global__ void deg_kernel(const int* __restrict__ dst, float* __restrict__ deg, int E)
{
    int e = blockIdx.x * blockDim.x + threadIdx.x;
    if (e < E) atomicAdd(&deg[dst[e]], 1.0f);
}

// Edge scatter-add of projected features P (cols 0..127 of proj[N x 256]).
__global__ void scatter_add(const float* __restrict__ proj,
                            const int* __restrict__ src,
                            const int* __restrict__ dst,
                            float* __restrict__ agg, int E)
{
    int t = blockIdx.x * blockDim.x + threadIdx.x;
    if (t >= E * 32) return;
    int e  = t >> 5;
    int c4 = (t & 31) << 2;
    int s = src[e], d = dst[e];
    float4 p = *(const float4*)(proj + (size_t)s * 256 + c4);
    float* o = agg + (size_t)d * HID + c4;
    atomicAdd(o + 0, p.x);
    atomicAdd(o + 1, p.y);
    atomicAdd(o + 2, p.z);
    atomicAdd(o + 3, p.w);
}

// h_pre = agg/max(deg,1) + b + Q  (Q = cols 128..255 of proj); accumulate
// per-channel sum / sumsq with one block-level partial -> 2 atomics/channel.
__global__ __launch_bounds__(HID)
void combine_bnstats(const float* __restrict__ agg,
                     const float* __restrict__ deg,
                     const float* __restrict__ proj,
                     const float* __restrict__ bvec,
                     float* __restrict__ hpre,
                     float* __restrict__ stats,  // [0..127]=sum, [128..255]=sumsq
                     int N, int rowsPerBlock)
{
    int c  = threadIdx.x;
    int r0 = blockIdx.x * rowsPerBlock;
    float b = bvec[c];
    float s = 0.f, s2 = 0.f;
    for (int i = 0; i < rowsPerBlock; ++i) {
        int r = r0 + i;
        if (r >= N) break;
        float d = deg[r];
        d = d > 1.f ? d : 1.f;
        float v = agg[(size_t)r * HID + c] / d + b
                + proj[(size_t)r * 256 + HID + c];
        hpre[(size_t)r * HID + c] = v;
        s  += v;
        s2 += v * v;
    }
    atomicAdd(&stats[c], s);
    atomicAdd(&stats[HID + c], s2);
}

// BN + ReLU, output packed bf16 [N x 64 dwords] (feeds next GEMM's A).
__global__ void bn_apply_bf16(const float* __restrict__ hpre,
                              unsigned int* __restrict__ h16,
                              const float* __restrict__ stats,
                              const float* __restrict__ gamma,
                              const float* __restrict__ beta, int N)
{
    int idx = blockIdx.x * blockDim.x + threadIdx.x;
    if (idx >= N * 64) return;
    int cd = idx & 63;
    int c0 = cd * 2;
    int r  = idx >> 6;
    float inv_n = 1.0f / (float)N;
    float m0 = stats[c0] * inv_n,     m1 = stats[c0 + 1] * inv_n;
    float v0 = stats[HID + c0] * inv_n - m0 * m0;
    float v1 = stats[HID + c0 + 1] * inv_n - m1 * m1;
    const float* hp = hpre + (size_t)r * HID + c0;
    float x0 = (hp[0] - m0) * rsqrtf(v0 + BN_EPS) * gamma[c0] + beta[c0];
    float x1 = (hp[1] - m1) * rsqrtf(v1 + BN_EPS) * gamma[c0 + 1] + beta[c0 + 1];
    x0 = fmaxf(x0, 0.f);
    x1 = fmaxf(x1, 0.f);
    h16[(size_t)r * 64 + cd] = pk2(x0, x1);
}

// BN (no ReLU), fp32 in place (final layer, feeds max-pool).
__global__ void bn_apply_f32(float* __restrict__ h,
                             const float* __restrict__ stats,
                             const float* __restrict__ gamma,
                             const float* __restrict__ beta, int N)
{
    int idx = blockIdx.x * blockDim.x + threadIdx.x;
    if (idx >= N * HID) return;
    int c = idx & (HID - 1);
    float inv_n = 1.0f / (float)N;
    float m = stats[c] * inv_n;
    float v = stats[HID + c] * inv_n - m * m;
    h[idx] = (h[idx] - m) * rsqrtf(v + BN_EPS) * gamma[c] + beta[c];
}

__device__ __forceinline__ void atomicMaxF(float* addr, float v)
{
    if (v >= 0.f)
        atomicMax((int*)addr, __float_as_int(v));
    else
        atomicMin((unsigned int*)addr, __float_as_uint(v));
}

__global__ void pool_init(float* __restrict__ pooled, int n)
{
    int i = blockIdx.x * blockDim.x + threadIdx.x;
    if (i < n) pooled[i] = -__builtin_inff();
}

__global__ void pool_max(const float* __restrict__ h,
                         const int* __restrict__ batch,
                         float* __restrict__ pooled, int N)
{
    int idx = blockIdx.x * blockDim.x + threadIdx.x;
    if (idx >= N * HID) return;
    int r = idx >> 7;
    int c = idx & (HID - 1);
    int g = batch[r];
    atomicMaxF(&pooled[(size_t)g * HID + c], h[idx]);
}

__global__ void final_linear(const float* __restrict__ pooled,
                             const float* __restrict__ Wlin,
                             const float* __restrict__ blin,
                             float* __restrict__ out, int G)
{
    int t = blockIdx.x * blockDim.x + threadIdx.x;
    if (t >= G * N_CLS) return;
    int g = t / N_CLS, j = t - g * N_CLS;
    float s = blin[j];
    const float* p = pooled + (size_t)g * HID;
    const float* w = Wlin + (size_t)j * HID;
#pragma unroll 4
    for (int c = 0; c < HID; ++c) s += p[c] * w[c];
    out[t] = s;
}

// ---------------------------------------------------------------------------
extern "C" void kernel_launch(void* const* d_in, const int* in_sizes, int n_in,
                              void* d_out, int out_size, void* d_ws, size_t ws_size,
                              hipStream_t stream)
{
    const float* x     = (const float*)d_in[0];
    const int*   ei    = (const int*)d_in[1];
    const int*   batch = (const int*)d_in[2];
    const float* W1l = (const float*)d_in[3];
    const float* b1  = (const float*)d_in[4];
    const float* W1r = (const float*)d_in[5];
    const float* g1  = (const float*)d_in[6];
    const float* be1 = (const float*)d_in[7];
    const float* W2l = (const float*)d_in[8];
    const float* b2  = (const float*)d_in[9];
    const float* W2r = (const float*)d_in[10];
    const float* g2  = (const float*)d_in[11];
    const float* be2 = (const float*)d_in[12];
    const float* W3l = (const float*)d_in[13];
    const float* b3  = (const float*)d_in[14];
    const float* W3r = (const float*)d_in[15];
    const float* g3  = (const float*)d_in[16];
    const float* be3 = (const float*)d_in[17];
    const float* Wlin = (const float*)d_in[18];
    const float* blin = (const float*)d_in[19];

    const int N = N_NODES, E = in_sizes[1] / 2, G = N_GRAPHS;
    const int* src = ei;
    const int* dst = ei + E;

    // workspace carve-out
    float* ws     = (float*)d_ws;
    float* proj   = ws;                          // N*256 f32
    float* hpre   = proj  + (size_t)N * 256;     // N*128 f32 (BN input; final h)
    float* agg    = hpre  + (size_t)N * HID;     // N*128 f32
    float* deg    = agg   + (size_t)N * HID;     // N
    float* stats  = deg   + N;                   // 256
    float* pooled = stats + 256;                 // G*128
    unsigned int* h16  = (unsigned int*)(pooled + (size_t)G * HID); // N*64 dwords
    unsigned int* wcat = h16 + (size_t)N * 64;   // 256*(768/2) dwords max

    const int gemmBlocks = (N + 63) / 64;
    const int combBlocks = (N + 255) / 256;
    const int elemBlocks = (N * HID + 255) / 256;
    const int bn16Blocks = (N * 64 + 255) / 256;
    const int scatBlocks = (E * 32 + 255) / 256;

    // degrees (same edge list for all layers)
    hipMemsetAsync(deg, 0, (size_t)N * sizeof(float), stream);
    deg_kernel<<<(E + 255) / 256, 256, 0, stream>>>(dst, deg, E);

    // ---- layer 1 (A = fp32 x) ----
    convert_w<<<(256 * (IN_DIM / 2) + 255) / 256, 256, 0, stream>>>(W1l, W1r, wcat, IN_DIM);
    gemm_dual<false><<<gemmBlocks, 256, 0, stream>>>(x, wcat, proj, N, IN_DIM);
    hipMemsetAsync(agg, 0, (size_t)N * HID * sizeof(float), stream);
    hipMemsetAsync(stats, 0, 256 * sizeof(float), stream);
    scatter_add<<<scatBlocks, 256, 0, stream>>>(proj, src, dst, agg, E);
    combine_bnstats<<<combBlocks, HID, 0, stream>>>(agg, deg, proj, b1, hpre, stats, N, 256);
    bn_apply_bf16<<<bn16Blocks, 256, 0, stream>>>(hpre, h16, stats, g1, be1, N);

    // ---- layer 2 (A = bf16 h) ----
    convert_w<<<(256 * (HID / 2) + 255) / 256, 256, 0, stream>>>(W2l, W2r, wcat, HID);
    gemm_dual<true><<<gemmBlocks, 256, 0, stream>>>(h16, wcat, proj, N, HID);
    hipMemsetAsync(agg, 0, (size_t)N * HID * sizeof(float), stream);
    hipMemsetAsync(stats, 0, 256 * sizeof(float), stream);
    scatter_add<<<scatBlocks, 256, 0, stream>>>(proj, src, dst, agg, E);
    combine_bnstats<<<combBlocks, HID, 0, stream>>>(agg, deg, proj, b2, hpre, stats, N, 256);
    bn_apply_bf16<<<bn16Blocks, 256, 0, stream>>>(hpre, h16, stats, g2, be2, N);

    // ---- layer 3 (A = bf16 h, no ReLU, fp32 out) ----
    convert_w<<<(256 * (HID / 2) + 255) / 256, 256, 0, stream>>>(W3l, W3r, wcat, HID);
    gemm_dual<true><<<gemmBlocks, 256, 0, stream>>>(h16, wcat, proj, N, HID);
    hipMemsetAsync(agg, 0, (size_t)N * HID * sizeof(float), stream);
    hipMemsetAsync(stats, 0, 256 * sizeof(float), stream);
    scatter_add<<<scatBlocks, 256, 0, stream>>>(proj, src, dst, agg, E);
    combine_bnstats<<<combBlocks, HID, 0, stream>>>(agg, deg, proj, b3, hpre, stats, N, 256);
    bn_apply_f32<<<elemBlocks, 256, 0, stream>>>(hpre, stats, g3, be3, N);

    // ---- global max pool + classifier ----
    pool_init<<<(G * HID + 255) / 256, 256, 0, stream>>>(pooled, G * HID);
    pool_max<<<elemBlocks, 256, 0, stream>>>(hpre, batch, pooled, N);
    final_linear<<<(G * N_CLS + 255) / 256, 256, 0, stream>>>(pooled, Wlin, blin,
                                                              (float*)d_out, G);
}
